// HydraFluxEncoder_23570780520938
// MI455X (gfx1250) — matrix operand order repaired and checked by reading, not compiled, not run on hardware
//
#include <hip/hip_runtime.h>
#include <hip/hip_bf16.h>
#include <math.h>

typedef __attribute__((ext_vector_type(16))) _Float16 v16h;
typedef __attribute__((ext_vector_type(8)))  _Float16 v8h;
typedef __attribute__((ext_vector_type(8)))  float    v8f;

constexpr int Bn = 16, Tn = 512, Dn = 1024, Hn = 8, Ln = 6, Fn = 4096;
constexpr int HDn = 128, BANDSn = 16;
constexpr int NBIN = Tn / 2 + 1;        // 257 rfft bins
constexpr int POOLROWS = 1920;          // sum over heads of T/res[h]

__constant__ int c_res[8]  = {1, 1, 2, 2, 4, 4, 8, 8};
__constant__ int c_hoff[8] = {0, 512, 1024, 1280, 1536, 1664, 1792, 1856};

// ---- CDNA5 async global->LDS copy (guarded; falls back to sync v8h copy) ---
#if defined(__gfx1250__) && __has_builtin(__builtin_amdgcn_global_load_async_to_lds_b128)
#define HAVE_ASYNC_LDS 1
#endif

__device__ __forceinline__ void copy8(const _Float16* g, _Float16* l) {
#ifdef HAVE_ASYNC_LDS
  typedef int vsi4 __attribute__((vector_size(16)));
  __builtin_amdgcn_global_load_async_to_lds_b128(
      (__attribute__((address_space(1))) vsi4*)g,
      (__attribute__((address_space(3))) vsi4*)l, 0, 0);
#else
  *(v8h*)l = *(const v8h*)g;
#endif
}

__device__ __forceinline__ void wait_async() {
#ifdef HAVE_ASYNC_LDS
#if __has_builtin(__builtin_amdgcn_s_wait_asynccnt)
  __builtin_amdgcn_s_wait_asynccnt(0);
#else
  asm volatile("s_wait_asynccnt 0x0" ::: "memory");
#endif
#endif
}

// ---------------------------------------------------------------------------
// x[b,t,:] = embed[tokens[b,t],:] + pos_embed[t,:]
// ---------------------------------------------------------------------------
__global__ void embed_kernel(const int* __restrict__ tok,
                             const float* __restrict__ emb,
                             const float* __restrict__ pos,
                             float* __restrict__ x) {
  int row = blockIdx.x;          // b*T + t
  int t   = row % Tn;
  int tk  = tok[row];
  const float* e = emb + (size_t)tk * Dn;
  const float* p = pos + (size_t)t * Dn;
  float* o = x + (size_t)row * Dn;
  for (int d = threadIdx.x; d < Dn; d += blockDim.x) o[d] = e[d] + p[d];
}

// ---------------------------------------------------------------------------
// LayerNorm over D=1024; optional f16 and/or f32 outputs.
// ---------------------------------------------------------------------------
__global__ __launch_bounds__(256)
void ln_kernel(const float* __restrict__ x, const float* __restrict__ g,
               const float* __restrict__ b, _Float16* __restrict__ o16,
               float* __restrict__ o32) {
  __shared__ float rs[256], rq[256];
  int row = blockIdx.x;
  const float* xr = x + (size_t)row * Dn;
  float s = 0.f, q = 0.f;
  float v[4];
#pragma unroll
  for (int i = 0; i < 4; ++i) {
    v[i] = xr[threadIdx.x + i * 256];
    s += v[i];
    q += v[i] * v[i];
  }
  rs[threadIdx.x] = s;
  rq[threadIdx.x] = q;
  __syncthreads();
  for (int st = 128; st > 0; st >>= 1) {
    if (threadIdx.x < st) {
      rs[threadIdx.x] += rs[threadIdx.x + st];
      rq[threadIdx.x] += rq[threadIdx.x + st];
    }
    __syncthreads();
  }
  float mean = rs[0] * (1.f / Dn);
  float var  = rq[0] * (1.f / Dn) - mean * mean;
  float rstd = rsqrtf(var + 1e-5f);
#pragma unroll
  for (int i = 0; i < 4; ++i) {
    int d = threadIdx.x + i * 256;
    float y = (v[i] - mean) * rstd * g[d] + b[d];
    if (o16) o16[(size_t)row * Dn + d] = (_Float16)y;
    if (o32) o32[(size_t)row * Dn + d] = y;
  }
}

// ---------------------------------------------------------------------------
// Transpose + f32->f16 convert:  Wt[N][K] (f16) = W[K][N] (f32).
// 32x32 tiles via LDS, 256 threads.
// ---------------------------------------------------------------------------
__global__ __launch_bounds__(256)
void convT_kernel(const float* __restrict__ W, _Float16* __restrict__ Wt,
                  int K, int N) {
  __shared__ float tile[32][33];
  int n0 = blockIdx.x * 32, k0 = blockIdx.y * 32;
  int tx = threadIdx.x & 31, ty = threadIdx.x >> 5;   // ty: 0..7
#pragma unroll
  for (int i = 0; i < 32; i += 8)
    tile[ty + i][tx] = W[(long long)(k0 + ty + i) * N + n0 + tx];
  __syncthreads();
#pragma unroll
  for (int i = 0; i < 32; i += 8)
    Wt[(long long)(n0 + ty + i) * K + k0 + tx] = (_Float16)tile[tx][ty + i];
}

// ---------------------------------------------------------------------------
// WMMA GEMM:  C = act(A16[MxK] * Bt16[NxK]^T + bias) (+res), out f32/f16.
// Block tile 128 x (NF*64), 8 waves (2x4), wave tile 64 x (NF*16), K-step 32.
// Both A and B are f16; B is pre-transposed ([N][K]) so staging is a pure
// global->LDS copy (async on CDNA5) and fragments are contiguous LDS reads
// matching the documented 16-bit WMMA A/B VGPR layouts.
// Per-matrix batch offset = (z/mod)*sq + (z%mod)*sr.
// ---------------------------------------------------------------------------
template <int NF>
__global__ __launch_bounds__(256)
void gemm_wmma(const _Float16* __restrict__ A, const _Float16* __restrict__ Bt,
               const float* __restrict__ bias, const float* __restrict__ res,
               float* __restrict__ outF, _Float16* __restrict__ outH,
               int M, int N, int K, int lda, int ldb, int ldc, int doGelu,
               int aMod, long long aSq, long long aSr,
               int bMod, long long bSq, long long bSr,
               int cMod, long long cSq, long long cSr) {
  constexpr int BN = NF * 64;
  __shared__ _Float16 sA[2][128 * 40];
  __shared__ _Float16 sB[2][BN * 40];

  int z = blockIdx.z;
  A  += (long long)(z / aMod) * aSq + (long long)(z % aMod) * aSr;
  Bt += (long long)(z / bMod) * bSq + (long long)(z % bMod) * bSr;
  long long coff = (long long)(z / cMod) * cSq + (long long)(z % cMod) * cSr;

  int n0   = blockIdx.x * BN;
  int m0   = blockIdx.y * 128;
  int tid  = threadIdx.x;
  int lane = tid & 31;
  int wave = tid >> 5;
  int wm   = wave >> 2;   // 0..1  (M, 64 rows each)
  int wn   = wave & 3;    // 0..3  (N, NF*16 cols each)

  v8f acc[4][NF];
#pragma unroll
  for (int i = 0; i < 4; ++i)
#pragma unroll
    for (int j = 0; j < NF; ++j) acc[i][j] = (v8f){0, 0, 0, 0, 0, 0, 0, 0};

  int nk = K / 32;

  auto stage = [&](int ks, int buf) {
    int k0 = ks * 32;
#pragma unroll
    for (int u = 0; u < 2; ++u) {               // A: 128 rows x 4 chunks(8h)
      int chunk = tid + u * 256;
      int row = chunk >> 2, seg = chunk & 3;
      copy8(A + (long long)(m0 + row) * lda + k0 + seg * 8,
            &sA[buf][row * 40 + seg * 8]);
    }
#pragma unroll
    for (int u = 0; u < NF; ++u) {              // B: BN rows x 4 chunks(8h)
      int chunk = tid + u * 256;
      int row = chunk >> 2, seg = chunk & 3;
      copy8(Bt + (long long)(n0 + row) * ldb + k0 + seg * 8,
            &sB[buf][row * 40 + seg * 8]);
    }
  };

  int r   = lane & 15;
  int hiL = lane >> 4;
  int kb  = hiL * 8;     // A K-base per lane half
  int khb = hiL * 8;     // B K-base per lane half

  stage(0, 0);
  for (int ks = 0; ks < nk; ++ks) {
    int cur = ks & 1;
    wait_async();
    __syncthreads();
    if (ks + 1 < nk) stage(ks + 1, cur ^ 1);
    if (ks + 2 < nk) {   // L2 warming via CDNA5 global_prefetch
      __builtin_prefetch(A + (long long)(m0 + (tid >> 2)) * lda + (ks + 2) * 32, 0, 1);
      __builtin_prefetch(Bt + (long long)(n0 + (tid >> 2)) * ldb + (ks + 2) * 32, 0, 1);
    }

    v16h afr[4];
#pragma unroll
    for (int fi = 0; fi < 4; ++fi) {
      const _Float16* ap = &sA[cur][(wm * 64 + fi * 16 + r) * 40];
      v8h lo = *(const v8h*)(ap + kb);
      v8h hi = *(const v8h*)(ap + kb + 16);
      afr[fi] = __builtin_shufflevector(lo, hi, 0, 1, 2, 3, 4, 5, 6, 7, 8, 9,
                                        10, 11, 12, 13, 14, 15);
    }
    v16h bfr[NF];
#pragma unroll
    for (int fj = 0; fj < NF; ++fj) {
      const _Float16* bp = &sB[cur][(wn * (NF * 16) + fj * 16 + r) * 40 + khb];
      v8h lo = *(const v8h*)(bp);
      v8h hi = *(const v8h*)(bp + 16);
      bfr[fj] = __builtin_shufflevector(lo, hi, 0, 1, 2, 3, 4, 5, 6, 7, 8, 9,
                                        10, 11, 12, 13, 14, 15);
    }
#pragma unroll
    for (int fi = 0; fi < 4; ++fi)
#pragma unroll
      for (int fj = 0; fj < NF; ++fj)
        acc[fi][fj] = __builtin_amdgcn_wmma_f32_16x16x32_f16(
            false, afr[fi], false, bfr[fj], (short)0, acc[fi][fj], false, false);
    __syncthreads();
  }

  // --- epilogue: bias / exact GELU / residual, f32 and/or f16 stores ---
  int rbase = hiL * 8;   // lanes 0-15 -> M rows i, lanes 16-31 -> rows i+8
#pragma unroll
  for (int fi = 0; fi < 4; ++fi) {
#pragma unroll
    for (int fj = 0; fj < NF; ++fj) {
      int gn = n0 + wn * (NF * 16) + fj * 16 + r;
      float bv = bias ? bias[gn] : 0.f;
#pragma unroll
      for (int i = 0; i < 8; ++i) {
        int gm = m0 + wm * 64 + fi * 16 + rbase + i;
        float vv = acc[fi][fj][i] + bv;
        if (doGelu) vv = 0.5f * vv * (1.f + erff(vv * 0.70710678118654752f));
        long long idx = coff + (long long)gm * ldc + gn;
        if (res)  vv += res[idx];
        if (outF) outF[idx] = vv;
        if (outH) outH[idx] = (_Float16)vv;
      }
    }
  }
}

// ---------------------------------------------------------------------------
// Pool K/V along T by the per-head resolution schedule (mean of r rows).
// ---------------------------------------------------------------------------
__global__ __launch_bounds__(256)
void pool_kernel(const float* __restrict__ k, const float* __restrict__ v,
                 float* __restrict__ kh, float* __restrict__ vh) {
  int bh = blockIdx.x;
  int b = bh / Hn, h = bh % Hn;
  int rr   = c_res[h];
  int Sh   = Tn / rr;
  int hoff = c_hoff[h];
  float inv = 1.f / rr;
  for (int idx = threadIdx.x; idx < Sh * HDn; idx += blockDim.x) {
    int s = idx >> 7;
    int c = idx & 127;
    float sk = 0.f, sv = 0.f;
    for (int j = 0; j < rr; ++j) {
      long long g = ((long long)(b * Tn + s * rr + j)) * Dn + h * HDn + c;
      sk += k[g];
      sv += v[g];
    }
    long long o = ((long long)(b * POOLROWS + hoff + s)) * HDn + c;
    kh[o] = sk * inv;
    vh[o] = sv * inv;
  }
}

// ---------------------------------------------------------------------------
// One (b,h,t) row per block: scores vs pooled keys, softmax, o = p @ vh.
// Output written f16, channel-major [b,h,c,t] so the spectral-filter GEMM
// can consume it through the transposed-B fast path.
// ---------------------------------------------------------------------------
__global__ __launch_bounds__(256)
void attn_kernel(const float* __restrict__ q, const float* __restrict__ kh,
                 const float* __restrict__ vh, _Float16* __restrict__ ot) {
  __shared__ float sc[Tn];
  __shared__ float red[256];
  __shared__ float qv[HDn];
  int blk = blockIdx.x;
  int t  = blk % Tn;
  int bh = blk / Tn;
  int h  = bh % Hn;
  int b  = bh / Hn;
  int rr = c_res[h], Sh = Tn / rr, hoff = c_hoff[h];
  const float* khb = kh + ((long long)(b * POOLROWS + hoff)) * HDn;
  const float* vhb = vh + ((long long)(b * POOLROWS + hoff)) * HDn;
  if (threadIdx.x < HDn)
    qv[threadIdx.x] = q[((long long)(b * Tn + t)) * Dn + h * HDn + threadIdx.x];
  __syncthreads();

  float lmax = -3.0e38f;
  for (int s = threadIdx.x; s < Sh; s += 256) {
    const float* kr = khb + (long long)s * HDn;
    float d = 0.f;
#pragma unroll 8
    for (int c = 0; c < HDn; ++c) d += qv[c] * kr[c];
    d *= 0.08838834764831845f;   // 1/sqrt(128)
    sc[s] = d;
    lmax = fmaxf(lmax, d);
  }
  red[threadIdx.x] = lmax;
  __syncthreads();
  for (int st = 128; st > 0; st >>= 1) {
    if (threadIdx.x < st)
      red[threadIdx.x] = fmaxf(red[threadIdx.x], red[threadIdx.x + st]);
    __syncthreads();
  }
  float mx = red[0];
  __syncthreads();

  float lsum = 0.f;
  for (int s = threadIdx.x; s < Sh; s += 256) {
    float e = expf(sc[s] - mx);
    sc[s] = e;
    lsum += e;
  }
  red[threadIdx.x] = lsum;
  __syncthreads();
  for (int st = 128; st > 0; st >>= 1) {
    if (threadIdx.x < st) red[threadIdx.x] += red[threadIdx.x + st];
    __syncthreads();
  }
  float inv = 1.f / red[0];

  if (threadIdx.x < HDn) {
    int c = threadIdx.x;
    float a = 0.f;
    for (int s = 0; s < Sh; ++s) a += sc[s] * vhb[(long long)s * HDn + c];
    ot[((long long)bh * HDn + c) * Tn + t] = (_Float16)(a * inv);
  }
}

// ---------------------------------------------------------------------------
// Time-domain filter kernel from band gates (gating real rfft gains ==
// circular convolution with a real, symmetric length-T kernel).
// ---------------------------------------------------------------------------
__global__ void buildw_kernel(const float* __restrict__ gates,
                              float* __restrict__ wbuf) {
  int h = blockIdx.x;
  const float* g = gates + h * BANDSn;
  for (int d = threadIdx.x; d < Tn; d += blockDim.x) {
    float acc = 0.f;
    for (int j = 0; j <= Tn / 2; ++j) {
      int band   = (j * BANDSn) / NBIN;
      float coef = (j == 0 || j == Tn / 2) ? 1.f : 2.f;
      acc += coef * g[band] *
             cosf(6.283185307179586f * (float)j * (float)d * (1.f / Tn));
    }
    wbuf[h * Tn + d] = acc * (1.f / Tn);
  }
}

// Expand per-head kernel into f16 circulant circ[h][t][s] = w[(t-s)%T].
__global__ void circ_kernel(const float* __restrict__ wbuf,
                            _Float16* __restrict__ circ) {
  int idx = blockIdx.x * blockDim.x + threadIdx.x;
  if (idx >= Hn * Tn * Tn) return;
  int s = idx & (Tn - 1);
  int t = (idx >> 9) & (Tn - 1);
  int h = idx >> 18;
  circ[idx] = (_Float16)wbuf[h * Tn + ((t - s) & (Tn - 1))];
}

// ---------------------------------------------------------------------------
extern "C" void kernel_launch(void* const* d_in, const int* in_sizes, int n_in,
                              void* d_out, int out_size, void* d_ws,
                              size_t ws_size, hipStream_t stream) {
  (void)in_sizes; (void)n_in; (void)out_size; (void)ws_size;
  const int*   tokens = (const int*)d_in[0];
  const float* embedw = (const float*)d_in[1];
  const float* posw   = (const float*)d_in[2];
  const float* Wq = (const float*)d_in[3];
  const float* bq = (const float*)d_in[4];
  const float* Wk = (const float*)d_in[5];
  const float* bk = (const float*)d_in[6];
  const float* Wv = (const float*)d_in[7];
  const float* bv = (const float*)d_in[8];
  const float* Wo = (const float*)d_in[9];
  const float* bo = (const float*)d_in[10];
  const float* ln1g = (const float*)d_in[11];
  const float* ln1b = (const float*)d_in[12];
  const float* ln2g = (const float*)d_in[13];
  const float* ln2b = (const float*)d_in[14];
  const float* W1 = (const float*)d_in[15];
  const float* b1 = (const float*)d_in[16];
  const float* W2 = (const float*)d_in[17];
  const float* b2 = (const float*)d_in[18];
  const float* gates = (const float*)d_in[19];
  const float* lnfg  = (const float*)d_in[20];
  const float* lnfb  = (const float*)d_in[21];

  char* wsp = (char*)d_ws;
  size_t off = 0;
  auto take = [&](size_t bytes) -> void* {
    void* p = wsp + off;
    off += (bytes + 255) & ~(size_t)255;
    return p;
  };
  const int MT = Bn * Tn;   // 8192 token rows
  float*    x     = (float*)take((size_t)MT * Dn * 4);
  _Float16* h16   = (_Float16*)take((size_t)MT * Dn * 2);
  float*    qb    = (float*)take((size_t)MT * Dn * 4);
  float*    kbuf  = (float*)take((size_t)MT * Dn * 4);
  float*    vbuf  = (float*)take((size_t)MT * Dn * 4);
  float*    khb   = (float*)take((size_t)Bn * POOLROWS * HDn * 4);
  float*    vhb   = (float*)take((size_t)Bn * POOLROWS * HDn * 4);
  _Float16* ot16  = (_Float16*)take((size_t)Bn * Hn * HDn * Tn * 2);
  _Float16* of16  = (_Float16*)take((size_t)MT * Dn * 2);
  _Float16* mid16 = (_Float16*)take((size_t)MT * Fn * 2);
  float*    wfil  = (float*)take((size_t)Hn * Tn * 4);
  _Float16* circ  = (_Float16*)take((size_t)Hn * Tn * Tn * 2);
  // per-layer transposed f16 weights (reused each layer)
  _Float16* wq16 = (_Float16*)take((size_t)Dn * Dn * 2);
  _Float16* wk16 = (_Float16*)take((size_t)Dn * Dn * 2);
  _Float16* wv16 = (_Float16*)take((size_t)Dn * Dn * 2);
  _Float16* wo16 = (_Float16*)take((size_t)Dn * Dn * 2);
  _Float16* w116 = (_Float16*)take((size_t)Dn * Fn * 2);
  _Float16* w216 = (_Float16*)take((size_t)Fn * Dn * 2);

  auto convT = [&](const float* W, _Float16* Wt, int K, int N) {
    convT_kernel<<<dim3(N / 32, K / 32), dim3(256), 0, stream>>>(W, Wt, K, N);
  };
  auto gemm = [&](int NF, const _Float16* A, const _Float16* Bt,
                  const float* bias, const float* res, float* oF,
                  _Float16* oH, int M, int N, int K, int lda, int ldb, int ldc,
                  int gelu, int batch,
                  int aM, long long aSq, long long aSr,
                  int bM, long long bSq, long long bSr,
                  int cM, long long cSq, long long cSr) {
    dim3 g(N / (NF * 64), M / 128, batch);
    if (NF == 4)
      gemm_wmma<4><<<g, dim3(256), 0, stream>>>(A, Bt, bias, res, oF, oH, M, N,
                                                K, lda, ldb, ldc, gelu, aM, aSq,
                                                aSr, bM, bSq, bSr, cM, cSq, cSr);
    else
      gemm_wmma<2><<<g, dim3(256), 0, stream>>>(A, Bt, bias, res, oF, oH, M, N,
                                                K, lda, ldb, ldc, gelu, aM, aSq,
                                                aSr, bM, bSq, bSr, cM, cSq, cSr);
  };

  embed_kernel<<<dim3(MT), dim3(256), 0, stream>>>(tokens, embedw, posw, x);

  for (int l = 0; l < Ln; ++l) {
    // one-time per layer: f32->f16 + transpose of all weights
    convT(Wq + (size_t)l * Dn * Dn, wq16, Dn, Dn);
    convT(Wk + (size_t)l * Dn * Dn, wk16, Dn, Dn);
    convT(Wv + (size_t)l * Dn * Dn, wv16, Dn, Dn);
    convT(Wo + (size_t)l * Dn * Dn, wo16, Dn, Dn);
    convT(W1 + (size_t)l * Dn * Fn, w116, Dn, Fn);
    convT(W2 + (size_t)l * Fn * Dn, w216, Fn, Dn);

    // --- attention block ---
    ln_kernel<<<dim3(MT), dim3(256), 0, stream>>>(
        x, ln1g + (size_t)l * Dn, ln1b + (size_t)l * Dn, h16, nullptr);
    gemm(4, h16, wq16, bq + (size_t)l * Dn, nullptr, qb, nullptr,
         MT, Dn, Dn, Dn, Dn, Dn, 0, 1, 1, 0, 0, 1, 0, 0, 1, 0, 0);
    gemm(4, h16, wk16, bk + (size_t)l * Dn, nullptr, kbuf, nullptr,
         MT, Dn, Dn, Dn, Dn, Dn, 0, 1, 1, 0, 0, 1, 0, 0, 1, 0, 0);
    gemm(4, h16, wv16, bv + (size_t)l * Dn, nullptr, vbuf, nullptr,
         MT, Dn, Dn, Dn, Dn, Dn, 0, 1, 1, 0, 0, 1, 0, 0, 1, 0, 0);
    pool_kernel<<<dim3(Bn * Hn), dim3(256), 0, stream>>>(kbuf, vbuf, khb, vhb);
    attn_kernel<<<dim3(Bn * Hn * Tn), dim3(256), 0, stream>>>(qb, khb, vhb,
                                                              ot16);

    // spectral band gating == per-(b,h) circulant matmul, done with WMMA
    buildw_kernel<<<dim3(Hn), dim3(256), 0, stream>>>(
        gates + (size_t)l * Hn * BANDSn, wfil);
    circ_kernel<<<dim3((Hn * Tn * Tn + 255) / 256), dim3(256), 0, stream>>>(
        wfil, circ);
    gemm(2, circ, ot16, nullptr, nullptr, nullptr, of16,
         Tn, HDn, Tn, Tn, Tn, Dn, 0, Bn * Hn,
         Hn, 0, (long long)Tn * Tn,                 // A: per-head circulant
         1, (long long)HDn * Tn, 0,                 // B: ot16[b,h] block
         Hn, (long long)Tn * Dn, (long long)HDn);   // C: of16 head slice

    gemm(4, of16, wo16, bo + (size_t)l * Dn, x, x, nullptr,
         MT, Dn, Dn, Dn, Dn, Dn, 0, 1, 1, 0, 0, 1, 0, 0, 1, 0, 0);

    // --- FFN block ---
    ln_kernel<<<dim3(MT), dim3(256), 0, stream>>>(
        x, ln2g + (size_t)l * Dn, ln2b + (size_t)l * Dn, h16, nullptr);
    gemm(4, h16, w116, b1 + (size_t)l * Fn, nullptr, nullptr, mid16,
         MT, Fn, Dn, Dn, Dn, Fn, 1, 1, 1, 0, 0, 1, 0, 0, 1, 0, 0);
    gemm(4, mid16, w216, b2 + (size_t)l * Dn, x, x, nullptr,
         MT, Dn, Fn, Fn, Fn, Dn, 0, 1, 1, 0, 0, 1, 0, 0, 1, 0, 0);
  }

  ln_kernel<<<dim3(MT), dim3(256), 0, stream>>>(x, lnfg, lnfb, nullptr,
                                                (float*)d_out);
}
